// RWKV_Tmix_x060_45732811768065
// MI455X (gfx1250) — compile-verified, matrix operand order checked by reading, not run on hardware
//
#include <hip/hip_runtime.h>
#include <hip/hip_bf16.h>

typedef __bf16 bf16;
typedef __attribute__((ext_vector_type(4)))  __bf16 bf16x4;
typedef __attribute__((ext_vector_type(8)))  __bf16 bf16x8;
typedef __attribute__((ext_vector_type(16))) __bf16 bf16x16;
typedef __attribute__((ext_vector_type(8)))  float  f32x8;

typedef __attribute__((ext_vector_type(4))) unsigned tdm_g0_t;
typedef __attribute__((ext_vector_type(8))) int      tdm_g1_t;
typedef __attribute__((ext_vector_type(4))) int      tdm_g23_t;
typedef __attribute__((ext_vector_type(8))) int      tdm_gx_t;

#if defined(__has_builtin)
#if __has_builtin(__builtin_amdgcn_tensor_load_to_lds)
#define HAVE_TDM 1
#endif
#endif
#ifndef HAVE_TDM
#define HAVE_TDM 0
#endif

#define B_SZ 4
#define T_SZ 2048
#define C_SZ 1024
#define H_SZ 64
#define M_SZ (B_SZ * T_SZ)   // 8192 rows

// ---------------------------------------------------------------- WMMA helpers

__device__ __forceinline__ f32x8 wmma_bf16f32(bf16x16 a, bf16x16 b, f32x8 c) {
  // D = A(16x32) * B(32x16) + C(16x16 f32)
  return __builtin_amdgcn_wmma_f32_16x16x32_bf16(false, a, false, b, (short)0, c,
                                                 false, false);
}

// 16-bit A/B fragment: lane holds 16 bf16.
// lanes 0-15 : K = {0..7, 16..23};  lanes 16-31 : K = {8..15, 24..31}
__device__ __forceinline__ bf16x16 frag_ld(const bf16* rowptr, int half) {
  bf16x8 lo = *(const bf16x8*)(rowptr + (half ? 8 : 0));
  bf16x8 hi = *(const bf16x8*)(rowptr + (half ? 24 : 16));
  return __builtin_shufflevector(lo, hi, 0, 1, 2, 3, 4, 5, 6, 7,
                                 8, 9, 10, 11, 12, 13, 14, 15);
}

// ------------------------------------ async global->LDS copy (16B per lane)
// LDS aperture keeps the wave-relative LDS address in addr[31:0], so the low
// 32 bits of a flat shared-memory pointer are the VDST LDS address.

__device__ __forceinline__ void async_copy16(void* lds, const void* gptr) {
  unsigned ldsoff = (unsigned)(unsigned long long)lds;
  asm volatile("global_load_async_to_lds_b128 %0, %1, off"
               :
               : "v"(ldsoff), "v"(gptr)
               : "memory");
}

__device__ __forceinline__ void async_wait0() {
  asm volatile("s_wait_asynccnt 0x0" ::: "memory");
}

#if HAVE_TDM
// ------------------------------------ TDM: 2D tile (tile_dim0=64 elems of 2B)
// D# per CDNA5 ISA. pad: every 32 DWORDs (128B row)
// insert 4 DWORDs (16B) -> LDS row stride 144B == LDS_P(72) bf16 elems.
__device__ __forceinline__ void tdm_load_2d(void* lds, const void* gsrc,
                                            int K, int tile_rows,
                                            int tensor_rows) {
  unsigned lds_addr = (unsigned)(unsigned long long)lds;
  unsigned long long ga = (unsigned long long)gsrc;
  tdm_g0_t g0;
  g0[0] = 1u;                                  // count=1 valid descriptor
  g0[1] = lds_addr;                            // lds_addr [63:32]
  g0[2] = (unsigned)ga;                        // global_addr lo
  g0[3] = (unsigned)(ga >> 32) | (2u << 30);   // global_addr hi | type=2
  tdm_g1_t g1;
  g1[0] = (1 << 16)      // data_size = 2 bytes
        | (1 << 20)      // pad_enable
        | (4 << 22)      // pad_interval: 32 DWORDs
        | (3 << 25);     // pad_amount: 4 DWORDs
  g1[1] = (int)((unsigned)(K & 0xFFFF) << 16);             // tensor_dim0 lo16
  g1[2] = (int)(((unsigned)K >> 16) |
                ((unsigned)(tensor_rows & 0xFFFF) << 16)); // dim0 hi | dim1 lo
  g1[3] = (int)(((unsigned)tensor_rows >> 16) | (64u << 16)); // dim1 hi|tile0=64
  g1[4] = tile_rows;                                       // tile_dim1 (tile2=0)
  g1[5] = K;                                               // dim0_stride lo32
  g1[6] = 0;
  g1[7] = 0;
  tdm_g23_t z4 = {0, 0, 0, 0};
  tdm_gx_t  z8 = {0, 0, 0, 0, 0, 0, 0, 0};
  // 6-arg form (clang-23 / therock-10.0 headers): groups 2,3 + extra group + cpol
  __builtin_amdgcn_tensor_load_to_lds(g0, g1, z4, z4, z8, 0);
}
#endif

// ---------------------------------------------------------------- prep: xx, xxx

__global__ void prep_kernel(const float* __restrict__ x,
                            const float* __restrict__ maa_x,
                            float* __restrict__ xx, bf16* __restrict__ xxx) {
  size_t idx = (size_t)blockIdx.x * 256 + threadIdx.x;
  int c = (int)(idx & (C_SZ - 1));
  size_t row = idx >> 10;
  int t = (int)(row & (T_SZ - 1));
  float xv = x[idx];
  float prev = (t > 0) ? x[idx - C_SZ] : 0.0f;
  float d = prev - xv;
  xx[idx] = d;
  xxx[idx] = (bf16)(xv + d * maa_x[c]);
}

// -------------------------------------------- transpose+convert weights to bf16

__global__ void transpose_bf16_kernel(const float* __restrict__ src,
                                      bf16* __restrict__ dst, int R, int Cc,
                                      int total) {
  int i = blockIdx.x * 256 + threadIdx.x;
  if (i >= total) return;
  int r = i / Cc;
  int c = i - r * Cc;
  dst[(size_t)c * R + r] = (bf16)src[i];  // dst is [Cc][R]
}

// -------------------------------------------- tiled bf16 WMMA GEMM
// C[M,N] = A[M,K] * BT[N,K]^T ; block tile 128x64x64, 8 waves, 2x2 WMMA / wave.
// A tile staged by TENSOR_LOAD_TO_LDS (TDM, hw row padding), B tile by
// GLOBAL_LOAD_ASYNC_TO_LDS_B128; double-buffered.

#define BM 128
#define BN 64
#define BK 64
#define LDS_P 72   // padded row (elems): 144B stride -> conflict-free frag reads

// EPI: 0 = f32 store, 1 = tanh -> bf16 store, 2 = exp(-exp(bias[col]+v)) f32
template <int EPI>
__global__ void __launch_bounds__(256)
gemm_tiled(const bf16* __restrict__ A, const bf16* __restrict__ BT,
           void* __restrict__ Out, const float* __restrict__ bias,
           int M, int N, int K) {
  __shared__ bf16 As[2][BM * LDS_P];
  __shared__ bf16 Bs[2][BN * LDS_P];

  const int tid = threadIdx.x;
  const int lane = tid & 31;
  const int wave = tid >> 5;
  const int half = lane >> 4;
  const int l15 = lane & 15;
  const int bm = blockIdx.x * BM;
  const int bn = blockIdx.y * BN;
  const int wm = (wave >> 1) * 32;  // 4 waves over M
  const int wn = (wave & 1) * 32;   // 2 waves over N

  f32x8 acc[2][2] = {};

  auto stage = [&](int buf, int k0) {
#if HAVE_TDM
    // A tile 128x64 via one TDM descriptor (wave 0 issues; EXEC ignored)
    if (wave == 0)
      tdm_load_2d(&As[buf][0], A + (size_t)bm * K + k0, K, BM, M);
#else
    // A tile: 128x64 bf16 = 1024 16B-chunks, 4 per thread (coalesced)
#pragma unroll
    for (int i = 0; i < 4; ++i) {
      int chunk = i * 256 + tid;
      int r = chunk >> 3;            // 0..127
      int c = (chunk & 7) * 8;       // 0..56
      async_copy16(&As[buf][r * LDS_P + c],
                   A + (size_t)(bm + r) * K + k0 + c);
    }
#endif
    // B tile: 64x64 bf16 = 512 chunks, 2 per thread
#pragma unroll
    for (int i = 0; i < 2; ++i) {
      int chunk = i * 256 + tid;
      int r = chunk >> 3;            // 0..63
      int c = (chunk & 7) * 8;
      async_copy16(&Bs[buf][r * LDS_P + c],
                   BT + (size_t)(bn + r) * K + k0 + c);
    }
  };

  auto stage_wait = [&]() {
#if HAVE_TDM
    __builtin_amdgcn_s_wait_tensorcnt(0);  // wave 0's TDM done (others: no-op)
#endif
    async_wait0();                         // my async B copies done
    __syncthreads();                       // all waves' copies visible
  };

  stage(0, 0);
  stage_wait();

  const int nk = K / BK;
  for (int kk = 0; kk < nk; ++kk) {
    int buf = kk & 1;
    if (kk + 1 < nk) stage(buf ^ 1, (kk + 1) * BK);
#pragma unroll
    for (int kc = 0; kc < 2; ++kc) {
#pragma unroll
      for (int i = 0; i < 2; ++i) {
        bf16x16 af =
            frag_ld(&As[buf][(wm + i * 16 + l15) * LDS_P + kc * 32], half);
#pragma unroll
        for (int j = 0; j < 2; ++j) {
          bf16x16 bfg =
              frag_ld(&Bs[buf][(wn + j * 16 + l15) * LDS_P + kc * 32], half);
          acc[i][j] = wmma_bf16f32(af, bfg, acc[i][j]);
        }
      }
    }
    stage_wait();  // copies into buf^1 landed (compute hid the latency)
  }

#pragma unroll
  for (int i = 0; i < 2; ++i) {
#pragma unroll
    for (int j = 0; j < 2; ++j) {
#pragma unroll
      for (int g = 0; g < 8; ++g) {
        int row = bm + wm + i * 16 + half * 8 + g;
        int col = bn + wn + j * 16 + l15;
        float v = acc[i][j][g];
        size_t idx = (size_t)row * N + col;
        if (EPI == 0) {
          ((float*)Out)[idx] = v;
        } else if (EPI == 1) {
          ((bf16*)Out)[idx] = (bf16)tanhf(v);
        } else {
          ((float*)Out)[idx] = __expf(-__expf(bias[col] + v));
        }
      }
    }
  }
}

// ------------------------------------- mix2: 4 rank-32 WMMA GEMMs + mix epilogue

__global__ void __launch_bounds__(256)
mix2_kernel(const bf16* __restrict__ mpre, const bf16* __restrict__ w2T,
            const float* __restrict__ x, const float* __restrict__ xx,
            const float* __restrict__ maa_w, const float* __restrict__ maa_k,
            const float* __restrict__ maa_v, const float* __restrict__ maa_r,
            bf16* __restrict__ ow, bf16* __restrict__ ok,
            bf16* __restrict__ ov, bf16* __restrict__ orr) {
  const int lane = threadIdx.x & 31;
  const int wave = threadIdx.x >> 5;
  const int tile = blockIdx.x * 8 + wave;  // 512*64 tiles
  const int tm = tile >> 6;
  const int tn = tile & 63;
  const int half = lane >> 4;
  const int l15 = lane & 15;

  f32x8 acc[4] = {};
#pragma unroll
  for (int f = 0; f < 4; ++f) {
    bf16x16 a = frag_ld(mpre + (size_t)(tm * 16 + l15) * 128 + f * 32, half);
    bf16x16 b = frag_ld(w2T + ((size_t)f * C_SZ + tn * 16 + l15) * 32, half);
    acc[f] = wmma_bf16f32(a, b, acc[f]);
  }

  const int col = tn * 16 + l15;
  const float mw = maa_w[col], mk = maa_k[col], mv = maa_v[col], mr = maa_r[col];
#pragma unroll
  for (int g = 0; g < 8; ++g) {
    int row = tm * 16 + half * 8 + g;
    size_t idx = (size_t)row * C_SZ + col;
    float xv = x[idx], xd = xx[idx];
    ow[idx] = (bf16)(xv + xd * (mw + acc[0][g]));
    ok[idx] = (bf16)(xv + xd * (mk + acc[1][g]));
    ov[idx] = (bf16)(xv + xd * (mv + acc[2][g]));
    orr[idx] = (bf16)(xv + xd * (mr + acc[3][g]));
  }
}

// ---------------------------------------------------------------- WKV6 scan
// one wave per (b,h), 8 waves/block; lane = (j, i-half); S[i][j] 8 regs/lane

__global__ void __launch_bounds__(256)
wkv_kernel(const float* __restrict__ r, const float* __restrict__ k,
           const float* __restrict__ v, const float* __restrict__ wd,
           const float* __restrict__ u, float* __restrict__ y) {
  const int id = blockIdx.x * 8 + (threadIdx.x >> 5);  // 0..255
  const int b = id >> 6;              // /H
  const int h = id & (H_SZ - 1);
  const int lane = threadIdx.x & 31;
  const int j = lane & 15;
  const int half = lane >> 4;
  const int i0 = half * 8;

  float S[8];
  float uv[8];
#pragma unroll
  for (int i = 0; i < 8; ++i) {
    S[i] = 0.0f;
    uv[i] = u[h * 16 + i0 + i];
  }

  for (int t = 0; t < T_SZ; ++t) {
    size_t base = ((size_t)(b * T_SZ + t)) * C_SZ + h * 16;
    if (t + 1 < T_SZ) {
      __builtin_prefetch(r + base + C_SZ, 0, 0);
      __builtin_prefetch(k + base + C_SZ, 0, 0);
      __builtin_prefetch(v + base + C_SZ, 0, 0);
      __builtin_prefetch(wd + base + C_SZ, 0, 0);
    }
    float rv[8], kv_[8], wv[8];
    *(float4*)&rv[0] = *(const float4*)(r + base + i0);
    *(float4*)&rv[4] = *(const float4*)(r + base + i0 + 4);
    *(float4*)&kv_[0] = *(const float4*)(k + base + i0);
    *(float4*)&kv_[4] = *(const float4*)(k + base + i0 + 4);
    *(float4*)&wv[0] = *(const float4*)(wd + base + i0);
    *(float4*)&wv[4] = *(const float4*)(wd + base + i0 + 4);
    float vj = v[base + j];

    float yp = 0.0f;
#pragma unroll
    for (int i = 0; i < 8; ++i) {
      float kvij = kv_[i] * vj;
      yp += rv[i] * (S[i] + uv[i] * kvij);
      S[i] = wv[i] * S[i] + kvij;
    }
    yp += __shfl_xor(yp, 16, 32);
    if (half == 0) y[base + j] = yp;
  }
}

// ---------------------------------------------------------------- LayerNorm

__global__ void __launch_bounds__(256)
ln_kernel(const float* __restrict__ y, const float* __restrict__ g,
          const float* __restrict__ bset, bf16* __restrict__ out) {
  const int wave = threadIdx.x >> 5;
  const int lane = threadIdx.x & 31;
  const size_t row = (size_t)blockIdx.x * 8 + wave;
  const float* p = y + row * C_SZ;

  float s = 0.0f, ss = 0.0f;
#pragma unroll
  for (int i = 0; i < 8; ++i) {
    float4 vv = *(const float4*)(p + i * 128 + lane * 4);
    s += vv.x + vv.y + vv.z + vv.w;
    ss += vv.x * vv.x + vv.y * vv.y + vv.z * vv.z + vv.w * vv.w;
  }
#pragma unroll
  for (int off = 16; off > 0; off >>= 1) {
    s += __shfl_xor(s, off, 32);
    ss += __shfl_xor(ss, off, 32);
  }
  const float mu = s * (1.0f / C_SZ);
  const float var = ss * (1.0f / C_SZ) - mu * mu;
  const float rstd = rsqrtf(var + 1e-5f);

#pragma unroll
  for (int i = 0; i < 8; ++i) {
    int c = i * 128 + lane * 4;
    float4 vv = *(const float4*)(p + c);
    bf16x4 o;
    o[0] = (bf16)((vv.x - mu) * rstd * g[c + 0] + bset[c + 0]);
    o[1] = (bf16)((vv.y - mu) * rstd * g[c + 1] + bset[c + 1]);
    o[2] = (bf16)((vv.z - mu) * rstd * g[c + 2] + bset[c + 2]);
    o[3] = (bf16)((vv.w - mu) * rstd * g[c + 3] + bset[c + 3]);
    *(bf16x4*)(out + row * C_SZ + c) = o;
  }
}

// ---------------------------------------------------------------- launch

extern "C" void kernel_launch(void* const* d_in, const int* in_sizes, int n_in,
                              void* d_out, int out_size, void* d_ws,
                              size_t ws_size, hipStream_t stream) {
  const float* x      = (const float*)d_in[0];
  const float* maa_x  = (const float*)d_in[1];
  const float* maa_w  = (const float*)d_in[2];
  const float* maa_k  = (const float*)d_in[3];
  const float* maa_v  = (const float*)d_in[4];
  const float* maa_r  = (const float*)d_in[5];
  const float* w1     = (const float*)d_in[6];   // [1024,128]
  const float* w2     = (const float*)d_in[7];   // [4,32,1024]
  const float* tdecay = (const float*)d_in[8];   // [1,1,1024]
  const float* tdw1   = (const float*)d_in[9];   // [1024,64]
  const float* tdw2   = (const float*)d_in[10];  // [64,1024]
  const float* faaaa  = (const float*)d_in[11];  // [64,16]
  const float* W_r    = (const float*)d_in[12];
  const float* W_k    = (const float*)d_in[13];
  const float* W_v    = (const float*)d_in[14];
  const float* W_o    = (const float*)d_in[15];
  const float* ln_g   = (const float*)d_in[16];
  const float* ln_b   = (const float*)d_in[17];

  char* ws = (char*)d_ws;
  size_t off = 0;
  auto alloc = [&](size_t bytes) {
    size_t o = off;
    off += (bytes + 255) & ~(size_t)255;
    return o;
  };
  const size_t MC = (size_t)M_SZ * C_SZ;

  float* xx    = (float*)(ws + alloc(MC * 4));
  bf16*  xxx   = (bf16*)(ws + alloc(MC * 2));   // reused later as y_ln
  bf16*  w1T   = (bf16*)(ws + alloc((size_t)128 * C_SZ * 2));
  bf16*  w2T   = (bf16*)(ws + alloc((size_t)4 * C_SZ * 32 * 2));
  bf16*  tdw1T = (bf16*)(ws + alloc((size_t)64 * C_SZ * 2));
  bf16*  tdw2T = (bf16*)(ws + alloc((size_t)C_SZ * 64 * 2));
  bf16*  WrT   = (bf16*)(ws + alloc((size_t)C_SZ * C_SZ * 2));
  bf16*  WkT   = (bf16*)(ws + alloc((size_t)C_SZ * C_SZ * 2));
  bf16*  WvT   = (bf16*)(ws + alloc((size_t)C_SZ * C_SZ * 2));
  bf16*  WoT   = (bf16*)(ws + alloc((size_t)C_SZ * C_SZ * 2));
  bf16*  mpre  = (bf16*)(ws + alloc((size_t)M_SZ * 128 * 2));
  bf16*  xw    = (bf16*)(ws + alloc(MC * 2));
  bf16*  xk    = (bf16*)(ws + alloc(MC * 2));
  bf16*  xvb   = (bf16*)(ws + alloc(MC * 2));
  bf16*  xr    = (bf16*)(ws + alloc(MC * 2));
  bf16*  hdec  = (bf16*)(ws + alloc((size_t)M_SZ * 64 * 2));
  float* wd    = (float*)(ws + alloc(MC * 4));
  float* rbuf  = (float*)(ws + alloc(MC * 4));
  float* kbuf  = (float*)(ws + alloc(MC * 4));
  float* vbuf  = (float*)(ws + alloc(MC * 4));
  float* ybuf  = xx;    // xx dead after mix2
  bf16*  yln   = xxx;   // xxx dead after mix1

  // 1) time-shift prep
  prep_kernel<<<(int)(MC / 256), 256, 0, stream>>>(x, maa_x, xx, xxx);

  // 2) weight convert+transpose to [N][K] bf16
  auto xpose = [&](const float* s, bf16* d, int R, int Cc) {
    int total = R * Cc;
    transpose_bf16_kernel<<<(total + 255) / 256, 256, 0, stream>>>(s, d, R, Cc,
                                                                   total);
  };
  xpose(w1, w1T, C_SZ, 128);
  for (int f = 0; f < 4; ++f)
    xpose(w2 + (size_t)f * 32 * C_SZ, w2T + (size_t)f * C_SZ * 32, 32, C_SZ);
  xpose(tdw1, tdw1T, C_SZ, 64);
  xpose(tdw2, tdw2T, 64, C_SZ);
  xpose(W_r, WrT, C_SZ, C_SZ);
  xpose(W_k, WkT, C_SZ, C_SZ);
  xpose(W_v, WvT, C_SZ, C_SZ);
  xpose(W_o, WoT, C_SZ, C_SZ);

  // 3) mix lora: mpre = tanh(xxx @ w1)   [8192,128]
  gemm_tiled<1><<<dim3(M_SZ / BM, 128 / BN), 256, 0, stream>>>(
      xxx, w1T, mpre, nullptr, M_SZ, 128, C_SZ);

  // 4) xw/xk/xv/xr = x + xx * (maa_* + mpre_f @ w2_f)
  mix2_kernel<<<(M_SZ / 16) * (C_SZ / 16) / 8, 256, 0, stream>>>(
      mpre, w2T, x, xx, maa_w, maa_k, maa_v, maa_r, xw, xk, xvb, xr);

  // 5) decay: hdec = tanh(xw @ tdw1); wd = exp(-exp(tdecay + hdec @ tdw2))
  gemm_tiled<1><<<dim3(M_SZ / BM, 64 / BN), 256, 0, stream>>>(
      xw, tdw1T, hdec, nullptr, M_SZ, 64, C_SZ);
  gemm_tiled<2><<<dim3(M_SZ / BM, C_SZ / BN), 256, 0, stream>>>(
      hdec, tdw2T, wd, tdecay, M_SZ, C_SZ, 64);

  // 6) r/k/v projections
  gemm_tiled<0><<<dim3(M_SZ / BM, C_SZ / BN), 256, 0, stream>>>(
      xr, WrT, rbuf, nullptr, M_SZ, C_SZ, C_SZ);
  gemm_tiled<0><<<dim3(M_SZ / BM, C_SZ / BN), 256, 0, stream>>>(
      xk, WkT, kbuf, nullptr, M_SZ, C_SZ, C_SZ);
  gemm_tiled<0><<<dim3(M_SZ / BM, C_SZ / BN), 256, 0, stream>>>(
      xvb, WvT, vbuf, nullptr, M_SZ, C_SZ, C_SZ);

  // 7) WKV6 recurrence (8 waves per block)
  wkv_kernel<<<(B_SZ * H_SZ) / 8, 256, 0, stream>>>(rbuf, kbuf, vbuf, wd, faaaa,
                                                    ybuf);

  // 8) LayerNorm -> bf16
  ln_kernel<<<M_SZ / 8, 256, 0, stream>>>(ybuf, ln_g, ln_b, yln);

  // 9) output projection -> d_out (f32)
  gemm_tiled<0><<<dim3(M_SZ / BM, C_SZ / BN), 256, 0, stream>>>(
      yln, WoT, (float*)d_out, nullptr, M_SZ, C_SZ, C_SZ);
}
